// Block2DAttention_30477087932845
// MI455X (gfx1250) — matrix operand order repaired and checked by reading, not compiled, not run on hardware
//
#include <hip/hip_runtime.h>

#define SH_ 2
#define SW_ 2

typedef __attribute__((ext_vector_type(16))) __bf16 v16bf;
typedef __attribute__((ext_vector_type(8)))  __bf16 v8bf;
typedef __attribute__((ext_vector_type(8)))  float  v8f;
typedef __attribute__((ext_vector_type(4)))  float  v4f;

__device__ __forceinline__ v8f vzero8() {
  v8f z;
#pragma unroll
  for (int i = 0; i < 8; ++i) z[i] = 0.0f;
  return z;
}

// ---- CDNA5 async global->LDS copy (ASYNCcnt-tracked), 16B per lane ----
__device__ __forceinline__ void async_ld16(void* lds_ptr, const void* gptr) {
  unsigned loff = (unsigned)(uintptr_t)lds_ptr;  // low 32 bits = LDS byte offset
  asm volatile("global_load_async_to_lds_b128 %0, %1, off"
               :: "v"(loff), "v"(gptr) : "memory");
}
__device__ __forceinline__ void wait_async0() {
  asm volatile("s_wait_asynccnt 0" ::: "memory");
}

// xor-butterfly across 16-lane groups, pure VALU (v_permlane16_b32)
__device__ __forceinline__ float plx16(float x, unsigned s0, unsigned s1) {
  return __uint_as_float(__builtin_amdgcn_permlane16(
      __float_as_uint(x), __float_as_uint(x), s0, s1, false, false));
}

// A-fragment for V_WMMA_F32_16X16X32_BF16: 16(M) x 32(K) from a row-major
// [16 x stride] LDS tile. Lanes 0-15: row=l, K {0..7}(v0-3) {16..23}(v4-7);
// lanes 16-31: row=l-16, K {8..15} {24..31}.  (ISA 7.12.2)
__device__ __forceinline__ v16bf frag_a(const __bf16* tile, int stride, int kofs) {
  const int lane = threadIdx.x & 31;
  const int r = lane & 15;
  const int h = lane >> 4;
  const __bf16* row = tile + r * stride + kofs + h * 8;
  v8bf lo = *(const v8bf*)(row);
  v8bf hi = *(const v8bf*)(row + 16);
  v16bf f;
#pragma unroll
  for (int i = 0; i < 8; ++i) { f[i] = lo[i]; f[i + 8] = hi[i]; }
  return f;
}

// B-fragment: 32(K) x 16(N), loaded from the row-major TRANSPOSE of B
// (a [16(N) x stride] tile). Lanes 0-15: col=l, K {0..7} {16..23}; lanes 16-31:
// col=l-16, K {8..15} {24..31} (same K order as frag_a).
__device__ __forceinline__ v16bf frag_b(const __bf16* tileT, int stride, int kofs) {
  const int lane = threadIdx.x & 31;
  const int r = lane & 15;
  const int h = lane >> 4;
  const __bf16* row = tileT + r * stride + kofs + h * 8;
  v8bf lo = *(const v8bf*)(row);
  v8bf hi = *(const v8bf*)(row + 16);
  v16bf f;
#pragma unroll
  for (int i = 0; i < 8; ++i) { f[i] = lo[i]; f[i + 8] = hi[i]; }
  return f;
}

// ---------------------------------------------------------------------------
// Shift + f32->bf16 convert of hidden_states (roll by (-SH,-SW) folded in).
// ---------------------------------------------------------------------------
__global__ __launch_bounds__(256) void shift_cvt(const float* __restrict__ x,
                                                 __bf16* __restrict__ xs) {
  const int row = blockIdx.x;            // 0..8191 (b*4096 + h*64 + w)
  const int col = threadIdx.x * 8;
  const int b = row >> 12;
  const int l = row & 4095;
  const int hsrc = ((l >> 6) + SH_) & 63;
  const int wsrc = ((l & 63) + SW_) & 63;
  const float* src = x + ((size_t)b * 4096 + hsrc * 64 + wsrc) * 2048 + col;
  v4f a = *(const v4f*)src;
  v4f c = *(const v4f*)(src + 4);
  v8bf r;
#pragma unroll
  for (int i = 0; i < 4; ++i) { r[i] = (__bf16)a[i]; r[i + 4] = (__bf16)c[i]; }
  *(v8bf*)(xs + (size_t)row * 2048 + col) = r;
}

__global__ __launch_bounds__(256) void cvt_bf16(const float* __restrict__ src,
                                                __bf16* __restrict__ dst, int n) {
  int i = (blockIdx.x * 256 + threadIdx.x) * 8;
  if (i + 8 > n) return;
  v4f a = *(const v4f*)(src + i);
  v4f c = *(const v4f*)(src + i + 4);
  v8bf r;
#pragma unroll
  for (int j = 0; j < 4; ++j) { r[j] = (__bf16)a[j]; r[j + 4] = (__bf16)c[j]; }
  *(v8bf*)(dst + i) = r;
}

// ---------------------------------------------------------------------------
// WMMA bf16 GEMM: C[8192,2048] = A[8192,2048] * B[2048,2048].
// Block tile 128x128, BK=64, LDS double-buffered (one barrier/iter).
// A-tile staged with GLOBAL_LOAD_ASYNC_TO_LDS_B128; B-tile transposed via regs.
// ---------------------------------------------------------------------------
template <typename OutT>
__global__ __launch_bounds__(256) void gemm_x_w(const __bf16* __restrict__ A,
                                                const __bf16* __restrict__ B,
                                                OutT* __restrict__ C) {
  __shared__ __align__(16) __bf16 As[2][128 * 72];   // [128 x 64] stride 72
  __shared__ __align__(16) __bf16 Bts[2][128 * 72];  // B^T [128(N) x 64(K)]

  const int t = threadIdx.x;
  const int lane = t & 31;
  const int wv = t >> 5;
  const int wm = wv & 1;
  const int wn = wv >> 1;
  const int m0 = blockIdx.y * 128;
  const int n0 = blockIdx.x * 128;

  v8f acc[4][2];
#pragma unroll
  for (int i = 0; i < 4; ++i)
#pragma unroll
    for (int j = 0; j < 2; ++j) acc[i][j] = vzero8();

  // ---- prologue: stage tile 0 into buffer 0 ----
  {
#pragma unroll
    for (int i = 0; i < 4; ++i) {
      int ch = i * 256 + t;
      int ar = ch >> 3, ak = (ch & 7) * 8;
      async_ld16(&As[0][ar * 72 + ak], A + (m0 + ar) * 2048 + ak);
      int bk = ch >> 4, bn = (ch & 15) * 8;
      v8bf bvv = *(const v8bf*)(B + bk * 2048 + n0 + bn);
#pragma unroll
      for (int j = 0; j < 8; ++j) Bts[0][(bn + j) * 72 + bk] = bvv[j];
    }
    wait_async0();
  }
  __syncthreads();

  int p = 0;
#pragma unroll 1
  for (int it = 0; it < 32; ++it) {
    const bool last = (it == 31);
    v8bf breg[4];
    if (!last) {
      const int kn = (it + 1) * 64;
#pragma unroll
      for (int i = 0; i < 4; ++i) {
        int ch = i * 256 + t;
        async_ld16(&As[p ^ 1][(ch >> 3) * 72 + (ch & 7) * 8],
                   A + (m0 + (ch >> 3)) * 2048 + kn + (ch & 7) * 8);
        breg[i] = *(const v8bf*)(B + (kn + (ch >> 4)) * 2048 + n0 + (ch & 15) * 8);
      }
    }

    // ---- compute on buffer p: 2 K-chunks x (4m x 2n) WMMAs ----
#pragma unroll
    for (int kc = 0; kc < 2; ++kc) {
      v16bf bfr[2];
#pragma unroll
      for (int nf = 0; nf < 2; ++nf)
        bfr[nf] = frag_b(&Bts[p][(wn * 32 + nf * 16) * 72], 72, kc * 32);
#pragma unroll
      for (int mf = 0; mf < 4; ++mf) {
        v16bf afr = frag_a(&As[p][(wm * 64 + mf * 16) * 72], 72, kc * 32);
#pragma unroll
        for (int nf = 0; nf < 2; ++nf)
          acc[mf][nf] = __builtin_amdgcn_wmma_f32_16x16x32_bf16(
              false, afr, false, bfr[nf], (short)0, acc[mf][nf], false, false);
      }
    }

    if (!last) {
#pragma unroll
      for (int i = 0; i < 4; ++i) {
        int ch = i * 256 + t;
        int bk = ch >> 4, bn = (ch & 15) * 8;
#pragma unroll
        for (int j = 0; j < 8; ++j) Bts[p ^ 1][(bn + j) * 72 + bk] = breg[i][j];
      }
      wait_async0();
      __syncthreads();
      p ^= 1;
    }
  }

  // ---- epilogue: int32 offsets, no branches ----
  const int r16 = lane & 15, hh = lane >> 4;
#pragma unroll
  for (int mf = 0; mf < 4; ++mf)
#pragma unroll
    for (int nf = 0; nf < 2; ++nf) {
      int off = (m0 + wm * 64 + mf * 16 + hh * 8) * 2048 +
                n0 + wn * 32 + nf * 16 + r16;
#pragma unroll
      for (int r = 0; r < 8; ++r) {
        C[off] = (OutT)acc[mf][nf][r];
        off += 2048;
      }
    }
}

// ---------------------------------------------------------------------------
// Flash-style block attention. One WG per (window, head, 128-row q-tile).
// 8 waves; each wave owns 16 q rows, streams 1024 keys in 32-key chunks.
// Q/K staged with async-to-LDS; K/V^T double-buffered (one barrier/iter).
// Row max via v_permlane16 butterflies; row sum via P x ones WMMA.
// Reverse cyclic shift folded into the output row index.
// ---------------------------------------------------------------------------
__global__ __launch_bounds__(256) void block_attn(
    const __bf16* __restrict__ q, const __bf16* __restrict__ k,
    const __bf16* __restrict__ v, __bf16* __restrict__ o) {
  __shared__ __align__(16) __bf16 Qs[128 * 72];       // [128 x 64] stride 72
  __shared__ __align__(16) __bf16 Ks[2][32 * 72];     // [32 keys x 64 hd]
  __shared__ __align__(16) __bf16 Vts[2][64 * 40];    // V^T [64 hd x 32 keys]
  __shared__ __align__(16) __bf16 Ps[8 * 16 * 40];    // per-wave P [16 x 32]

  const int t = threadIdx.x;
  const int lane = t & 31;
  const int wv = t >> 5;
  const int r16 = lane & 15, hh = lane >> 4;

  const int bid = blockIdx.x;
  const int blk = bid >> 8;           // 0..7 windows
  const int head = (bid >> 3) & 31;
  const int qt = bid & 7;
  const int batch = blk >> 2;
  const int bx = (blk >> 1) & 1;
  const int by = blk & 1;

  const int kr = t >> 3, koff = (t & 7) * 8;

  // stage Q tile (128 x 64) + K/V tile 0
#pragma unroll
  for (int c = 0; c < 4; ++c) {
    int ch = c * 256 + t;
    int qr = ch >> 3, off = (ch & 7) * 8;
    int pp = qt * 128 + qr;
    int grow = batch * 4096 + (bx * 32 + (pp >> 5)) * 64 + (by * 32 + (pp & 31));
    async_ld16(&Qs[qr * 72 + off], q + grow * 2048 + head * 64 + off);
  }
  {
    int grow = batch * 4096 + (bx * 32 + (kr >> 5)) * 64 + (by * 32 + (kr & 31));
    async_ld16(&Ks[0][kr * 72 + koff], k + grow * 2048 + head * 64 + koff);
    v8bf vd = *(const v8bf*)(v + grow * 2048 + head * 64 + koff);
#pragma unroll
    for (int j = 0; j < 8; ++j) Vts[0][(koff + j) * 40 + kr] = vd[j];
  }
  wait_async0();
  __syncthreads();

  const v16bf qf0 = frag_a(Qs + (wv * 16) * 72, 72, 0);
  const v16bf qf1 = frag_a(Qs + (wv * 16) * 72, 72, 32);

  v16bf ones;
#pragma unroll
  for (int i = 0; i < 16; ++i) ones[i] = (__bf16)1.0f;

  float m_i[8], l_i[8];
#pragma unroll
  for (int r = 0; r < 8; ++r) { m_i[r] = -3.0e38f; l_i[r] = 0.0f; }
  v8f oacc[4];
#pragma unroll
  for (int g = 0; g < 4; ++g) oacc[g] = vzero8();

  int p = 0;
#pragma unroll 1
  for (int kt = 0; kt < 32; ++kt) {
    const bool last = (kt == 31);
    v8bf vd;
    if (!last) {
      int pp = (kt + 1) * 32 + kr;
      int grow = batch * 4096 + (bx * 32 + (pp >> 5)) * 64 + (by * 32 + (pp & 31));
      async_ld16(&Ks[p ^ 1][kr * 72 + koff], k + grow * 2048 + head * 64 + koff);
      vd = *(const v8bf*)(v + grow * 2048 + head * 64 + koff);
    }

    // ---- S = Q K^T (16x32 per wave), HD=64 => 2 WMMAs per 16-key group ----
    v8f s[2];
#pragma unroll
    for (int st = 0; st < 2; ++st) {
      v16bf kb0 = frag_b(&Ks[p][(st * 16) * 72], 72, 0);
      v16bf kb1 = frag_b(&Ks[p][(st * 16) * 72], 72, 32);
      v8f a = vzero8();
      a = __builtin_amdgcn_wmma_f32_16x16x32_bf16(false, qf0, false, kb0,
                                                  (short)0, a, false, false);
      a = __builtin_amdgcn_wmma_f32_16x16x32_bf16(false, qf1, false, kb1,
                                                  (short)0, a, false, false);
      s[st] = a;
    }

    // ---- online softmax; row = r + 8*hh matches the C-fragment layout ----
    float alpha[8];
#pragma unroll
    for (int r = 0; r < 8; ++r) {
      float s0 = s[0][r] * 0.125f;          // 1/sqrt(64)
      float s1 = s[1][r] * 0.125f;
      float pm = fmaxf(s0, s1);
      pm = fmaxf(pm, plx16(pm, 0x67452301u, 0xEFCDAB89u));  // xor 1
      pm = fmaxf(pm, plx16(pm, 0x54761032u, 0xDCFE98BAu));  // xor 2
      pm = fmaxf(pm, plx16(pm, 0x32107654u, 0xBA98FEDCu));  // xor 4
      pm = fmaxf(pm, plx16(pm, 0xFEDCBA98u, 0x76543210u));  // xor 8
      float mn = fmaxf(m_i[r], pm);
      alpha[r] = __expf(m_i[r] - mn);
      s[0][r] = __expf(s0 - mn);
      s[1][r] = __expf(s1 - mn);
      m_i[r] = mn;
    }
#pragma unroll
    for (int g = 0; g < 4; ++g)
#pragma unroll
      for (int r = 0; r < 8; ++r) oacc[g][r] *= alpha[r];

    // ---- P -> per-wave LDS (bf16), reload as A-fragment ----
    __bf16* pw = Ps + wv * 16 * 40;
#pragma unroll
    for (int r = 0; r < 8; ++r) {
      pw[(r + hh * 8) * 40 + r16]      = (__bf16)s[0][r];
      pw[(r + hh * 8) * 40 + 16 + r16] = (__bf16)s[1][r];
    }
    asm volatile("s_wait_dscnt 0" ::: "memory");
    v16bf pf = frag_a(pw, 40, 0);

    // row sums of P via WMMA with all-ones B (every column = row sum)
    v8f ls = __builtin_amdgcn_wmma_f32_16x16x32_bf16(
        false, pf, false, ones, (short)0, vzero8(), false, false);
#pragma unroll
    for (int r = 0; r < 8; ++r) l_i[r] = l_i[r] * alpha[r] + ls[r];

#pragma unroll
    for (int g = 0; g < 4; ++g) {
      v16bf vf = frag_b(&Vts[p][(g * 16) * 40], 40, 0);
      oacc[g] = __builtin_amdgcn_wmma_f32_16x16x32_bf16(
          false, pf, false, vf, (short)0, oacc[g], false, false);
    }

    if (!last) {
#pragma unroll
      for (int j = 0; j < 8; ++j) Vts[p ^ 1][(koff + j) * 40 + kr] = vd[j];
      wait_async0();
      __syncthreads();
      p ^= 1;
    }
  }

  // ---- normalize + write with reverse cyclic shift folded in ----
#pragma unroll
  for (int r = 0; r < 8; ++r) {
    float inv = 1.0f / l_i[r];
    int prow = qt * 128 + wv * 16 + r + hh * 8;
    int h2 = (bx * 32 + (prow >> 5) + SH_) & 63;
    int w2 = (by * 32 + (prow & 31) + SW_) & 63;
    int off = (batch * 4096 + h2 * 64 + w2) * 2048 + head * 64 + r16;
#pragma unroll
    for (int g = 0; g < 4; ++g)
      o[off + g * 16] = (__bf16)(oacc[g][r] * inv);
  }
}

extern "C" void kernel_launch(void* const* d_in, const int* in_sizes, int n_in,
                              void* d_out, int out_size, void* d_ws, size_t ws_size,
                              hipStream_t stream) {
  (void)in_sizes; (void)n_in; (void)out_size; (void)ws_size;
  const float* hidden = (const float*)d_in[0];
  const float* Wq = (const float*)d_in[1];
  const float* Wk = (const float*)d_in[2];
  const float* Wv = (const float*)d_in[3];
  const float* Wo = (const float*)d_in[4];
  float* out = (float*)d_out;

  char* ws = (char*)d_ws;
  const size_t XS_BYTES = (size_t)8192 * 2048 * 2;  // 32 MiB (bf16 activations)
  const size_t W_BYTES  = (size_t)2048 * 2048 * 2;  // 8 MiB per weight
  __bf16* xs  = (__bf16*)(ws);
  __bf16* wqb = (__bf16*)(ws + XS_BYTES);
  __bf16* wkb = (__bf16*)(ws + XS_BYTES + 1 * W_BYTES);
  __bf16* wvb = (__bf16*)(ws + XS_BYTES + 2 * W_BYTES);
  __bf16* wob = (__bf16*)(ws + XS_BYTES + 3 * W_BYTES);
  __bf16* qb  = (__bf16*)(ws + 1 * XS_BYTES + 4 * W_BYTES);
  __bf16* kb  = (__bf16*)(ws + 2 * XS_BYTES + 4 * W_BYTES);
  __bf16* vb  = (__bf16*)(ws + 3 * XS_BYTES + 4 * W_BYTES);
  __bf16* ob  = (__bf16*)(ws + 4 * XS_BYTES + 4 * W_BYTES);

  shift_cvt<<<8192, 256, 0, stream>>>(hidden, xs);
  const int WN = 2048 * 2048;
  cvt_bf16<<<WN / (256 * 8), 256, 0, stream>>>(Wq, wqb, WN);
  cvt_bf16<<<WN / (256 * 8), 256, 0, stream>>>(Wk, wkb, WN);
  cvt_bf16<<<WN / (256 * 8), 256, 0, stream>>>(Wv, wvb, WN);
  cvt_bf16<<<WN / (256 * 8), 256, 0, stream>>>(Wo, wob, WN);

  dim3 gg(2048 / 128, 8192 / 128);
  gemm_x_w<__bf16><<<gg, 256, 0, stream>>>(xs, wqb, qb);
  gemm_x_w<__bf16><<<gg, 256, 0, stream>>>(xs, wkb, kb);
  gemm_x_w<__bf16><<<gg, 256, 0, stream>>>(xs, wvb, vb);

  block_attn<<<2048, 256, 0, stream>>>(qb, kb, vb, ob);

  gemm_x_w<float><<<gg, 256, 0, stream>>>(ob, wob, out);
}
